// MHA_3023656976407
// MI455X (gfx1250) — matrix operand order repaired and checked by reading, not compiled, not run on hardware
//
#include <hip/hip_runtime.h>
#include <hip/hip_bf16.h>

// ---------------------------------------------------------------------------
// Problem constants (from the reference)
// ---------------------------------------------------------------------------
#define B_  2
#define S_  2048
#define E_  2048
#define H_  16
#define KV_ 4
#define D_  128
#define M_  (B_ * S_)          // 4096 rows of x
#define NQKV_ 3072             // H*D + 2*KV*D

typedef __attribute__((ext_vector_type(16))) __bf16 v16bf;
typedef __attribute__((ext_vector_type(8)))  __bf16 v8bf;
typedef __attribute__((ext_vector_type(4)))  __bf16 v4bf;
typedef __attribute__((ext_vector_type(8)))  float  v8f;
typedef int v4i __attribute__((vector_size(16)));

union V16U { v16bf v; v8bf h[2]; };

// ---------------------------------------------------------------------------
// CDNA5 async global->LDS copy (GLOBAL_LOAD_ASYNC_TO_LDS_B128, ASYNCcnt).
// Probe-discovered signature: (v4i* gaddr, v4i* ldsaddr, int offset, int cpol).
// Guarded: falls back to a register-staged LDS copy if the builtin is absent.
// ---------------------------------------------------------------------------
#if __has_builtin(__builtin_amdgcn_global_load_async_to_lds_b128)
#define ASYNC_LDS_AVAILABLE 1
#else
#define ASYNC_LDS_AVAILABLE 0
#endif

static __device__ __forceinline__ void async_copy16(const __bf16* g, __bf16* l) {
#if ASYNC_LDS_AVAILABLE
  __builtin_amdgcn_global_load_async_to_lds_b128((v4i*)g, (v4i*)l, 0, 0);
#else
  *(v8bf*)l = *(const v8bf*)g;
#endif
}

static __device__ __forceinline__ void wait_async0() {
#if ASYNC_LDS_AVAILABLE
#if __has_builtin(__builtin_amdgcn_s_wait_asynccnt)
  __builtin_amdgcn_s_wait_asynccnt(0);
#else
  asm volatile("s_wait_asynccnt 0" ::: "memory");
#endif
#endif
}

// ---------------------------------------------------------------------------
// WMMA helpers — layouts used here (wave32); A/B k order is contiguous, not the ISA table order
// A 16x32 bf16: lane = {row = l&15, hi = l>>4}; elems 0..15 = K[hi*16 .. +15]
//               (two 16B loads)
// B 32x16 bf16: lane = {col = l&15, hi = l>>4}; elems 0..15 = K[hi*16 .. +15]
//               (one 32B load from a row-major [N,K] operand)
// C/D 16x16 f32: vgpr j: row = j + 8*(l>>4), col = l&15
// ---------------------------------------------------------------------------
static __device__ __forceinline__ v16bf load_a_frag(const __bf16* base, int ld, int lane) {
  const int row = lane & 15, hi = lane >> 4;
  const __bf16* p = base + (size_t)row * ld + hi * 16;
  V16U r;
  r.h[0] = *(const v8bf*)(p);
  r.h[1] = *(const v8bf*)(p + 8);
  return r.v;
}

static __device__ __forceinline__ v16bf load_b_frag(const __bf16* base, int ld, int lane) {
  const int col = lane & 15, hi = lane >> 4;
  return *(const v16bf*)(base + (size_t)col * ld + hi * 16);
}

static __device__ __forceinline__ v8f wmma_bf16(v16bf a, v16bf b, v8f c) {
  return __builtin_amdgcn_wmma_f32_16x16x32_bf16(
      /*neg_a=*/false, a, /*neg_b=*/false, b,
      /*c_mod=*/(short)0, c, /*reuse_a=*/false, /*reuse_b=*/false);
}

// ---------------------------------------------------------------------------
// Kernel 0: f32 -> bf16 conversion (grid-stride)
// ---------------------------------------------------------------------------
__global__ __launch_bounds__(256) void cvt_bf16_kernel(const float* __restrict__ in,
                                                       __bf16* __restrict__ out, long n) {
  long i = (long)blockIdx.x * blockDim.x + threadIdx.x;
  const long stride = (long)gridDim.x * blockDim.x;
  for (; i < n; i += stride) out[i] = (__bf16)in[i];
}

// ---------------------------------------------------------------------------
// Kernel 1/4: bf16 GEMM, C[m,n] = sum_k A[m,k] * W[n,k]   (both row-major, "TN")
// WG = 256 threads = 8 waves laid out 4(M) x 2(N); per-wave tile 32x64
// => 128x128 per workgroup, 8 wmma per 32-K step per wave.
// Operands double-buffered in LDS (2 x (8KB A + 8KB B)) via async copies:
// each 32-K step moves exactly 16KB/WG from L2 instead of ~48KB of redundant
// per-wave fetches. M,N multiples of 128; K multiple of 32.
// ---------------------------------------------------------------------------
__global__ __launch_bounds__(256) void gemm_bf16_kernel(
    const __bf16* __restrict__ A, const __bf16* __restrict__ W,
    float* __restrict__ C, int K, int ldc) {
  __shared__ __bf16 lds_a[2][128][32];
  __shared__ __bf16 lds_b[2][128][32];

  const int tid  = threadIdx.x;
  const int lane = tid & 31;
  const int wave = tid >> 5;
  const int wm = wave & 3, wn = wave >> 2;
  const int tm0 = blockIdx.y * 128;
  const int tn0 = blockIdx.x * 128;

  // cooperative tile staging: 256 threads x 2 x 16B each for A and for B
  const int crow = tid >> 1;        // 0..127 tile row
  const int cseg = (tid & 1) * 8;   // element offset 0 or 8; +16 covers rest

  auto stage = [&](int buf, int k0) {
    const __bf16* ga = A + (size_t)(tm0 + crow) * K + k0 + cseg;
    const __bf16* gb = W + (size_t)(tn0 + crow) * K + k0 + cseg;
    async_copy16(ga,      &lds_a[buf][crow][cseg]);
    async_copy16(ga + 16, &lds_a[buf][crow][cseg + 16]);
    async_copy16(gb,      &lds_b[buf][crow][cseg]);
    async_copy16(gb + 16, &lds_b[buf][crow][cseg + 16]);
  };

  stage(0, 0);
  wait_async0();
  __syncthreads();

  v8f acc[2][4] = {};
  int buf = 0;
  for (int k0 = 0; k0 < K; k0 += 32) {
    if (k0 + 32 < K) stage(buf ^ 1, k0 + 32);   // overlap next-tile DMA

    const __bf16* la = &lds_a[buf][wm * 32][0];
    v16bf a0 = load_a_frag(la, 32, lane);
    v16bf a1 = load_a_frag(la + 16 * 32, 32, lane);
#pragma unroll
    for (int nf = 0; nf < 4; ++nf) {
      v16bf b = load_b_frag(&lds_b[buf][wn * 64 + nf * 16][0], 32, lane);
      acc[0][nf] = wmma_bf16(a0, b, acc[0][nf]);
      acc[1][nf] = wmma_bf16(a1, b, acc[1][nf]);
    }

    wait_async0();      // our async stores into buf^1 have landed in LDS
    __syncthreads();    // everyone done reading buf, buf^1 fully staged
    buf ^= 1;
  }

  const int col = lane & 15, hi = lane >> 4;
#pragma unroll
  for (int mf = 0; mf < 2; ++mf)
#pragma unroll
    for (int nf = 0; nf < 4; ++nf)
#pragma unroll
      for (int j = 0; j < 8; ++j)
        C[(size_t)(tm0 + wm * 32 + mf * 16 + j + 8 * hi) * ldc +
          (tn0 + wn * 64 + nf * 16 + col)] = acc[mf][nf][j];
}

// ---------------------------------------------------------------------------
// Kernel 2: per-(b,s,unit) RMSNorm + RoPE + layout scatter. One wave (32
// threads) per unit; units 0..15 = Q heads, 16..19 = K heads, 20..23 = V heads.
// Q -> bf16 [B][H][S][D]; K -> bf16 [B][KV][S][D]; V -> bf16 transposed
// [B][KV][D][S] so the P@V WMMA B-fragments are contiguous loads.
// ---------------------------------------------------------------------------
__global__ __launch_bounds__(32) void norm_rope_scatter_kernel(
    const float* __restrict__ qkv, const float* __restrict__ cosb,
    const float* __restrict__ sinb, const float* __restrict__ qw,
    const float* __restrict__ kw, __bf16* __restrict__ qo,
    __bf16* __restrict__ ko, __bf16* __restrict__ vt) {
  const int lane = threadIdx.x;
  const int idx = blockIdx.x;
  const int unit = idx % 24;
  const long bs = idx / 24;
  const int b = (int)(bs / S_), s = (int)(bs % S_);

  const float* row = qkv + (size_t)bs * NQKV_;
  const float* src;
  if (unit < 16)      src = row + unit * D_;
  else if (unit < 20) src = row + H_ * D_ + (unit - 16) * D_;
  else                src = row + H_ * D_ + KV_ * D_ + (unit - 20) * D_;

  const int d0 = lane * 4;
  float4 xv = *(const float4*)(src + d0);
  float x[4] = {xv.x, xv.y, xv.z, xv.w};

  if (unit < 20) {
    // RMSNorm over the 128-wide head (full-wave butterfly reduce)
    float ss = x[0] * x[0] + x[1] * x[1] + x[2] * x[2] + x[3] * x[3];
    ss += __shfl_xor(ss, 1, 32);
    ss += __shfl_xor(ss, 2, 32);
    ss += __shfl_xor(ss, 4, 32);
    ss += __shfl_xor(ss, 8, 32);
    ss += __shfl_xor(ss, 16, 32);
    const float r = rsqrtf(ss * (1.0f / D_) + 1e-6f);
    const float* w = (unit < 16) ? qw : kw;
    float y[4], part[4], o[4];
#pragma unroll
    for (int i = 0; i < 4; ++i) y[i] = x[i] * r * w[d0 + i];
    // RoPE: pair (d, d+64); d and d^64 live in lanes l and l^16
#pragma unroll
    for (int i = 0; i < 4; ++i) part[i] = __shfl_xor(y[i], 16, 32);
    const float* cs = cosb + (size_t)s * (D_ / 2);
    const float* sn = sinb + (size_t)s * (D_ / 2);
#pragma unroll
    for (int i = 0; i < 4; ++i) {
      const int d = d0 + i;
      if (d < D_ / 2) o[i] = y[i] * cs[d] - part[i] * sn[d];
      else            o[i] = part[i] * sn[d - D_ / 2] + y[i] * cs[d - D_ / 2];
    }
    __bf16* dst;
    if (unit < 16) dst = qo + (((size_t)b * H_ + unit) * S_ + s) * D_;
    else           dst = ko + (((size_t)b * KV_ + (unit - 16)) * S_ + s) * D_;
    v4bf ov;
#pragma unroll
    for (int i = 0; i < 4; ++i) ov[i] = (__bf16)o[i];
    *(v4bf*)(dst + d0) = ov;
  } else {
    // V: no norm/rope; store transposed [kv][d][s]
    const int kv = unit - 20;
    __bf16* dst = vt + ((size_t)b * KV_ + kv) * D_ * S_;
#pragma unroll
    for (int i = 0; i < 4; ++i) dst[(size_t)(d0 + i) * S_ + s] = (__bf16)x[i];
  }
}

// ---------------------------------------------------------------------------
// Kernel 3: causal flash attention (GQA).
// 256-thread WG = 8 independent waves; each wave owns 16 query rows,
// streams keys 32 at a time. Per block:
//   S = Q K^T : 8 wmma (D=128 -> 4 K-steps x 2 key tiles)
//   online softmax in f32 (butterfly max/sum over the 16-lane half-groups)
//   P -> LDS bf16 (C-layout -> A-layout re-shaping), s_wait_dscnt 0
//   O += P V : 8 wmma against Vt (B-frags contiguous)
// No barriers: waves have per-wave causal trip counts; EXEC stays full.
// ---------------------------------------------------------------------------
__global__ __launch_bounds__(256) void flash_attn_kernel(
    const __bf16* __restrict__ Q, const __bf16* __restrict__ Kt,
    const __bf16* __restrict__ Vt, __bf16* __restrict__ ctx) {
  __shared__ __bf16 lds_p[8][16][32];

  const int lane = threadIdx.x & 31;
  const int wave = threadIdx.x >> 5;
  const int qb = blockIdx.x & 15;
  const int h  = (blockIdx.x >> 4) & 15;
  const int b  = blockIdx.x >> 8;
  const int qr = qb * 128 + wave * 16;   // first query row for this wave
  const int kvh = h >> 2;                // GQA: 4 query heads per kv head

  const __bf16* Qb = Q  + (((size_t)b * H_ + h) * S_ + qr) * D_;
  const __bf16* Kb = Kt + (((size_t)b * KV_ + kvh) * S_) * D_;
  const __bf16* Vb = Vt + (((size_t)b * KV_ + kvh) * D_) * S_;

  const int col = lane & 15, hi = lane >> 4;

  v16bf qa[4];
#pragma unroll
  for (int dk = 0; dk < 4; ++dk) qa[dk] = load_a_frag(Qb + dk * 32, D_, lane);

  v8f o[8] = {};
  float m_r[8], l_r[8];
#pragma unroll
  for (int j = 0; j < 8; ++j) { m_r[j] = -1e30f; l_r[j] = 0.0f; }

  const float scale = 0.0883883476483184f;  // 1/sqrt(128)
  const int kend = qr + 15;                 // last key needed by this wave

  for (int kk = 0; kk <= kend; kk += 32) {
    // ---- scores for keys [kk, kk+32) ----
    v8f s0 = {}, s1 = {};
#pragma unroll
    for (int dk = 0; dk < 4; ++dk) {
      v16bf b0 = load_b_frag(Kb + (size_t)kk * D_ + dk * 32, D_, lane);
      v16bf b1 = load_b_frag(Kb + (size_t)(kk + 16) * D_ + dk * 32, D_, lane);
      s0 = wmma_bf16(qa[dk], b0, s0);
      s1 = wmma_bf16(qa[dk], b1, s1);
    }

    // ---- online softmax update ----
    float alpha[8];
#pragma unroll
    for (int j = 0; j < 8; ++j) {
      const int rowg = qr + j + 8 * hi;
      const bool ok0 = (kk + col) <= rowg;
      const bool ok1 = (kk + 16 + col) <= rowg;
      float sv0 = ok0 ? s0[j] * scale : -1e30f;
      float sv1 = ok1 ? s1[j] * scale : -1e30f;
      float mx = fmaxf(sv0, sv1);
      mx = fmaxf(mx, __shfl_xor(mx, 1, 32));
      mx = fmaxf(mx, __shfl_xor(mx, 2, 32));
      mx = fmaxf(mx, __shfl_xor(mx, 4, 32));
      mx = fmaxf(mx, __shfl_xor(mx, 8, 32));
      const float mnew = fmaxf(m_r[j], mx);
      const float p0 = ok0 ? __expf(sv0 - mnew) : 0.0f;
      const float p1 = ok1 ? __expf(sv1 - mnew) : 0.0f;
      float ps = p0 + p1;
      ps += __shfl_xor(ps, 1, 32);
      ps += __shfl_xor(ps, 2, 32);
      ps += __shfl_xor(ps, 4, 32);
      ps += __shfl_xor(ps, 8, 32);
      const float a = __expf(m_r[j] - mnew);
      l_r[j] = l_r[j] * a + ps;
      m_r[j] = mnew;
      alpha[j] = a;
      lds_p[wave][j + 8 * hi][col]      = (__bf16)p0;
      lds_p[wave][j + 8 * hi][col + 16] = (__bf16)p1;
    }

    // rescale output accumulators
#pragma unroll
    for (int t = 0; t < 8; ++t)
#pragma unroll
      for (int j = 0; j < 8; ++j) o[t][j] *= alpha[j];

    asm volatile("s_wait_dscnt 0" ::: "memory");  // P tile visible to our DS reads

    // ---- O += P @ V ----
    v16bf pa = load_a_frag(&lds_p[wave][0][0], 32, lane);
#pragma unroll
    for (int t = 0; t < 8; ++t) {
      v16bf vb = load_b_frag(Vb + (size_t)(t * 16) * S_ + kk, S_, lane);
      o[t] = wmma_bf16(pa, vb, o[t]);
    }
  }

  // ---- normalize and write ctx[b*S+row][h*D + d] as bf16 ----
  float inv_l[8];
#pragma unroll
  for (int j = 0; j < 8; ++j) inv_l[j] = 1.0f / l_r[j];
  __bf16* cbase = ctx + ((size_t)(b * S_ + qr)) * (H_ * D_) + h * D_;
#pragma unroll
  for (int t = 0; t < 8; ++t)
#pragma unroll
    for (int j = 0; j < 8; ++j)
      cbase[(size_t)(j + 8 * hi) * (H_ * D_) + t * 16 + col] =
          (__bf16)(o[t][j] * inv_l[j]);
}

// ---------------------------------------------------------------------------
// Orchestration
// ---------------------------------------------------------------------------
extern "C" void kernel_launch(void* const* d_in, const int* in_sizes, int n_in,
                              void* d_out, int out_size, void* d_ws, size_t ws_size,
                              hipStream_t stream) {
  (void)in_sizes; (void)n_in; (void)out_size; (void)ws_size;
  const float* x    = (const float*)d_in[0];
  // d_in[1] = attention_mask (causal; handled analytically)
  const float* cosb = (const float*)d_in[2];
  const float* sinb = (const float*)d_in[3];
  const float* wq   = (const float*)d_in[4];
  const float* wk   = (const float*)d_in[5];
  const float* wv   = (const float*)d_in[6];
  const float* wo   = (const float*)d_in[7];
  const float* qnw  = (const float*)d_in[8];
  const float* knw  = (const float*)d_in[9];
  float* out = (float*)d_out;

  // workspace carve-up (~124 MB total)
  char* ws = (char*)d_ws;
  size_t off = 0;
  auto carve = [&](size_t bytes) -> char* {
    char* p = ws + off;
    off += (bytes + 255) & ~(size_t)255;
    return p;
  };
  __bf16* x_bf   = (__bf16*)carve((size_t)M_ * E_ * 2);
  __bf16* wq_bf  = (__bf16*)carve((size_t)H_ * D_ * E_ * 2);
  __bf16* wk_bf  = (__bf16*)carve((size_t)KV_ * D_ * E_ * 2);
  __bf16* wv_bf  = (__bf16*)carve((size_t)KV_ * D_ * E_ * 2);
  __bf16* wo_bf  = (__bf16*)carve((size_t)E_ * H_ * D_ * 2);
  float*  qkv    = (float*) carve((size_t)M_ * NQKV_ * 4);
  __bf16* q_bf   = (__bf16*)carve((size_t)B_ * H_ * S_ * D_ * 2);
  __bf16* k_bf   = (__bf16*)carve((size_t)B_ * KV_ * S_ * D_ * 2);
  __bf16* vt_bf  = (__bf16*)carve((size_t)B_ * KV_ * D_ * S_ * 2);
  __bf16* ctx_bf = (__bf16*)carve((size_t)M_ * H_ * D_ * 2);

  // 1) f32 -> bf16 converts
  cvt_bf16_kernel<<<2048, 256, 0, stream>>>(x,  x_bf,  (long)M_ * E_);
  cvt_bf16_kernel<<<2048, 256, 0, stream>>>(wq, wq_bf, (long)H_ * D_ * E_);
  cvt_bf16_kernel<<<1024, 256, 0, stream>>>(wk, wk_bf, (long)KV_ * D_ * E_);
  cvt_bf16_kernel<<<1024, 256, 0, stream>>>(wv, wv_bf, (long)KV_ * D_ * E_);
  cvt_bf16_kernel<<<2048, 256, 0, stream>>>(wo, wo_bf, (long)E_ * (H_ * D_));

  // 2) QKV projections into one [M, 3072] f32 buffer (ldc = 3072)
  gemm_bf16_kernel<<<dim3((H_ * D_) / 128, M_ / 128), 256, 0, stream>>>(
      x_bf, wq_bf, qkv, E_, NQKV_);
  gemm_bf16_kernel<<<dim3((KV_ * D_) / 128, M_ / 128), 256, 0, stream>>>(
      x_bf, wk_bf, qkv + H_ * D_, E_, NQKV_);
  gemm_bf16_kernel<<<dim3((KV_ * D_) / 128, M_ / 128), 256, 0, stream>>>(
      x_bf, wv_bf, qkv + H_ * D_ + KV_ * D_, E_, NQKV_);

  // 3) RMSNorm + RoPE + scatter to attention layouts
  norm_rope_scatter_kernel<<<B_ * S_ * (H_ + 2 * KV_), 32, 0, stream>>>(
      qkv, cosb, sinb, qnw, knw, q_bf, k_bf, vt_bf);

  // 4) causal flash attention -> ctx bf16 [M, H*D]
  flash_attn_kernel<<<B_ * H_ * (S_ / 128), 256, 0, stream>>>(
      q_bf, k_bf, vt_bf, ctx_bf);

  // 5) output projection -> f32 d_out [M, E]
  gemm_bf16_kernel<<<dim3(E_ / 128, M_ / 128), 256, 0, stream>>>(
      ctx_bf, wo_bf, out, H_ * D_, E_);
}
